// GPT_11098195493117
// MI455X (gfx1250) — matrix operand order repaired and checked by reading, not compiled, not run on hardware
//
#include <hip/hip_runtime.h>
#include <cstdint>
#include <cstddef>

// GPT-2 small forward on gfx1250: bf16 WMMA (V_WMMA_F32_16X16X32_BF16) for all
// GEMMs + flash-attention; fp32 accumulate. A-tiles move via
// GLOBAL_LOAD_ASYNC_TO_LDS_B128 (ASYNCcnt), B-tiles prefetched with
// global_prefetch and converted fp32->bf16 with packed cvt into transposed LDS.

#define E_  768
#define H_  12
#define HS_ 64
#define L_  12
#define T_  1024
#define B_  4
#define M_  (B_ * T_)      // 4096 rows
#define FF_ 3072
#define V_  50257

typedef __attribute__((ext_vector_type(16))) __bf16 v16bf;
typedef __attribute__((ext_vector_type(8)))  float  v8f;

union Frag {
  v16bf v;
  uint4 u[2];
  unsigned short s[16];
};

__device__ __forceinline__ unsigned short f2bf(float f) {
  unsigned int u = __float_as_uint(f);
  u += 0x7FFFu + ((u >> 16) & 1u);       // round-to-nearest-even
  return (unsigned short)(u >> 16);
}

// pack two fp32 -> packed bf16x2 (lo = first arg)
__device__ __forceinline__ unsigned pack_bf16x2(float lo, float hi) {
#if __has_builtin(__builtin_amdgcn_cvt_pk_bf16_f32)
  union { __attribute__((ext_vector_type(2))) __bf16 v; unsigned u; } t;
  t.v = __builtin_amdgcn_cvt_pk_bf16_f32(lo, hi);
  return t.u;
#else
  return ((unsigned)f2bf(hi) << 16) | (unsigned)f2bf(lo);
#endif
}

__device__ __forceinline__ v8f wmma_bf16(const Frag& a, const Frag& b, v8f c) {
  // (neg_a, A, neg_b, B, c_mod, C, reuse_a, reuse_b)
  return __builtin_amdgcn_wmma_f32_16x16x32_bf16(false, a.v, false, b.v,
                                                 (short)0, c, false, false);
}

// ---------------------------------------------------------------- embedding
__global__ __launch_bounds__(256) void embed_kernel(
    const int* __restrict__ idx, const float* __restrict__ tok_w,
    const float* __restrict__ pos_w, float* __restrict__ x) {
  int row = blockIdx.x;              // b*T + t
  int t = row & (T_ - 1);
  int tok = idx[row];
  const float* tw = tok_w + (size_t)tok * E_;
  const float* pw = pos_w + (size_t)t * E_;
  float* xr = x + (size_t)row * E_;
  for (int e = threadIdx.x; e < E_; e += 256) xr[e] = tw[e] + pw[e];
}

// ---------------------------------------------------------------- layernorm
__global__ __launch_bounds__(256) void ln_kernel(
    const float* __restrict__ x, const float* __restrict__ w,
    const float* __restrict__ bia, unsigned short* __restrict__ out) {
  int row = blockIdx.x;
  int tid = threadIdx.x;
  const float* xr = x + (size_t)row * E_;
  float v0 = xr[tid], v1 = xr[tid + 256], v2 = xr[tid + 512];
  float s  = v0 + v1 + v2;
  float sq = v0 * v0 + v1 * v1 + v2 * v2;
  #pragma unroll
  for (int off = 16; off; off >>= 1) {
    s  += __shfl_xor(s, off);
    sq += __shfl_xor(sq, off);
  }
  __shared__ float ss[8], ssq[8];
  if ((tid & 31) == 0) { ss[tid >> 5] = s; ssq[tid >> 5] = sq; }
  __syncthreads();
  float ts = 0.f, tsq = 0.f;
  #pragma unroll
  for (int i = 0; i < 8; ++i) { ts += ss[i]; tsq += ssq[i]; }
  float mean = ts * (1.f / E_);
  float var  = tsq * (1.f / E_) - mean * mean;
  float rstd = rsqrtf(var + 1e-5f);
  unsigned short* orow = out + (size_t)row * E_;
  orow[tid]       = f2bf((v0 - mean) * rstd * w[tid]       + bia[tid]);
  orow[tid + 256] = f2bf((v1 - mean) * rstd * w[tid + 256] + bia[tid + 256]);
  orow[tid + 512] = f2bf((v2 - mean) * rstd * w[tid + 512] + bia[tid + 512]);
}

// ---------------------------------------------------------------- GEMM
// C[M,N] = A[M,K](bf16) * B[K,N](f32 -> bf16 in LDS)
// Block: 128x128, 8 waves (2x4), wave tile 64x32, BK=32.
// A tile: GLOBAL_LOAD_ASYNC_TO_LDS_B128 (no VGPR round-trip, ASYNCcnt).
// B tile: prefetch next tile; packed cvt into transposed Bt[n][k].
template<bool OUT_BF16, bool BIAS, bool RES, bool RELU>
__global__ __launch_bounds__(256) void gemm_kernel(
    const unsigned short* __restrict__ A, const float* __restrict__ Bm,
    const float* __restrict__ bias, const float* __restrict__ res,
    void* __restrict__ Cout, int M, int N, int K) {
  __shared__ unsigned short As[128][40];   // row-major, 80B rows (16B-aligned chunks)
  __shared__ unsigned short Bt[128][40];   // transposed [n][k]

  const int tid  = threadIdx.x;
  const int lane = tid & 31;
  const int wid  = tid >> 5;
  const int wm = wid >> 2, wn = wid & 3;
  const int lm = lane & 15, hv = lane >> 4;
  const int blockM = blockIdx.y << 7, blockN = blockIdx.x << 7;

  const v8f vzero = {0.f, 0.f, 0.f, 0.f, 0.f, 0.f, 0.f, 0.f};
  v8f acc[4][2];
  #pragma unroll
  for (int i = 0; i < 4; ++i)
    #pragma unroll
    for (int j = 0; j < 2; ++j) acc[i][j] = vzero;

  const int arow = tid >> 1, akc = (tid & 1) << 4;   // A: 128 rows x 32 k
  const int bk2 = (tid & 15) << 1;                   // B: k pair 0..30
  const int bn0 = (tid >> 4) << 3;                   // B: n chunk of 8

  const unsigned ldsA = (unsigned)(uintptr_t)&As[arow][akc];
  unsigned* const ldsBrowBase = (unsigned*)&Bt[0][bk2];

  for (int kt = 0; kt < K; kt += 32) {
    // ---- A tile: async DMA global->LDS (bf16, 2 x 16B per lane)
    {
      const unsigned short* gA = A + (size_t)(blockM + arow) * K + kt + akc;
      asm volatile(
          "global_load_async_to_lds_b128 %0, %1, off\n\t"
          "global_load_async_to_lds_b128 %0, %1, off offset:16"
          :: "v"(ldsA), "v"(gA) : "memory");
    }
    // ---- B tile: 2(k) x 8(n) patch per thread, packed bf16 stores
    {
      const float* b0 = Bm + (size_t)(kt + bk2) * N + blockN + bn0;
      const float* b1 = b0 + N;
      if (kt + 32 < K) {
        __builtin_prefetch(b0 + 32 * (size_t)N, 0, 1);   // global_prefetch
        __builtin_prefetch(b1 + 32 * (size_t)N, 0, 1);
      }
      #pragma unroll
      for (int ii = 0; ii < 8; ++ii) {
        int gn = blockN + bn0 + ii;
        float f0 = (gn < N) ? b0[ii] : 0.f;
        float f1 = (gn < N) ? b1[ii] : 0.f;
        // Bt[n][k] row stride 40 elems = 20 dwords; bk2 even -> dword aligned
        ldsBrowBase[(size_t)(bn0 + ii) * 20] = pack_bf16x2(f0, f1);
      }
    }
    asm volatile("s_wait_asynccnt 0x0" ::: "memory");
    __syncthreads();

    Frag af[4], bf[2];
    #pragma unroll
    for (int i = 0; i < 4; ++i) {   // A frag: lane m=lm, k chunks 8*hv, 16+8*hv
      int r = (wm << 6) + (i << 4) + lm;
      af[i].u[0] = *(const uint4*)&As[r][hv << 3];
      af[i].u[1] = *(const uint4*)&As[r][16 + (hv << 3)];
    }
    #pragma unroll
    for (int j = 0; j < 2; ++j) {   // B frag: lane n=lm, k chunks 8*hv, 16+8*hv
      int n = (wn << 5) + (j << 4) + lm;
      bf[j].u[0] = *(const uint4*)&Bt[n][hv << 3];
      bf[j].u[1] = *(const uint4*)&Bt[n][16 + (hv << 3)];
    }
    #pragma unroll
    for (int i = 0; i < 4; ++i)
      #pragma unroll
      for (int j = 0; j < 2; ++j)
        acc[i][j] = wmma_bf16(af[i], bf[j], acc[i][j]);
    __syncthreads();
  }

  // epilogue: C layout => row = r + 8*hv, col = lm (per 16x16 tile)
  #pragma unroll
  for (int i = 0; i < 4; ++i) {
    #pragma unroll
    for (int j = 0; j < 2; ++j) {
      int gn = blockN + (wn << 5) + (j << 4) + lm;
      if (gn < N) {
        #pragma unroll
        for (int r = 0; r < 8; ++r) {
          int gm = blockM + (wm << 6) + (i << 4) + r + (hv << 3);
          float vv = acc[i][j][r];
          if constexpr (BIAS) vv += bias[gn];
          if constexpr (RES)  vv += res[(size_t)gm * N + gn];
          if constexpr (RELU) vv = fmaxf(vv, 0.f);
          if constexpr (OUT_BF16)
            ((unsigned short*)Cout)[(size_t)gm * N + gn] = f2bf(vv);
          else
            ((float*)Cout)[(size_t)gm * N + gn] = vv;
        }
      }
    }
  }
}

// ---------------------------------------------------------------- attention
// One wave32 per (16 query rows, head, batch). Flash-style online softmax.
__global__ __launch_bounds__(32) void attn_kernel(
    const unsigned short* __restrict__ q, const unsigned short* __restrict__ k,
    const unsigned short* __restrict__ v, unsigned short* __restrict__ o) {
  __shared__ unsigned short Plds[16][40];
  const int lane = threadIdx.x;
  const int lm = lane & 15, hv = lane >> 4;
  const int q0 = blockIdx.x << 4;
  const int h  = blockIdx.y;
  const int b  = blockIdx.z;
  const size_t rowbase = (size_t)(b * T_) * E_ + (size_t)h * HS_;

  Frag aQ0, aQ1;                         // Q rows q0..q0+15, d = 0..63
  {
    const unsigned short* qr = q + rowbase + (size_t)(q0 + lm) * E_;
    aQ0.u[0] = *(const uint4*)(qr + (hv << 3));
    aQ0.u[1] = *(const uint4*)(qr + 16 + (hv << 3));
    aQ1.u[0] = *(const uint4*)(qr + 32 + (hv << 3));
    aQ1.u[1] = *(const uint4*)(qr + 48 + (hv << 3));
  }

  const v8f vzero = {0.f, 0.f, 0.f, 0.f, 0.f, 0.f, 0.f, 0.f};
  float mrow[8], lsum[8];
  v8f oacc[4];
  #pragma unroll
  for (int r = 0; r < 8; ++r) { mrow[r] = -1e30f; lsum[r] = 0.f; }
  #pragma unroll
  for (int jd = 0; jd < 4; ++jd) oacc[jd] = vzero;

  const float scale = 0.125f;            // HS^-0.5

  for (int s0 = 0; s0 < q0 + 16; s0 += 32) {
    v8f sc[2] = {vzero, vzero};
    #pragma unroll
    for (int js = 0; js < 2; ++js) {     // S = Q * K^T  (B frag lane = key row)
      const unsigned short* kr = k + rowbase + (size_t)(s0 + (js << 4) + lm) * E_;
      Frag b0, b1;
      b0.u[0] = *(const uint4*)(kr + (hv << 3));
      b0.u[1] = *(const uint4*)(kr + 16 + (hv << 3));
      b1.u[0] = *(const uint4*)(kr + 32 + (hv << 3));
      b1.u[1] = *(const uint4*)(kr + 48 + (hv << 3));
      sc[js] = wmma_bf16(aQ0, b0, sc[js]);
      sc[js] = wmma_bf16(aQ1, b1, sc[js]);
    }
    #pragma unroll
    for (int r = 0; r < 8; ++r) {        // online softmax per owned row
      int rowg = q0 + r + (hv << 3);
      float s0v = sc[0][r] * scale;
      float s1v = sc[1][r] * scale;
      if (s0 + lm > rowg)      s0v = -1e30f;
      if (s0 + 16 + lm > rowg) s1v = -1e30f;
      float bm = fmaxf(s0v, s1v);
      #pragma unroll
      for (int off = 8; off; off >>= 1) bm = fmaxf(bm, __shfl_xor(bm, off));
      float nm = fmaxf(mrow[r], bm);
      float al = __expf(mrow[r] - nm);
      float p0 = __expf(s0v - nm);
      float p1 = __expf(s1v - nm);
      float rs = p0 + p1;
      #pragma unroll
      for (int off = 8; off; off >>= 1) rs += __shfl_xor(rs, off);
      lsum[r] = lsum[r] * al + rs;
      mrow[r] = nm;
      #pragma unroll
      for (int jd = 0; jd < 4; ++jd) oacc[jd][r] *= al;
      Plds[r + (hv << 3)][lm]      = f2bf(p0);
      Plds[r + (hv << 3)][lm + 16] = f2bf(p1);
    }
    __syncthreads();                     // C-layout -> A-layout via LDS
    Frag pa;
    pa.u[0] = *(const uint4*)&Plds[lm][hv << 3];
    pa.u[1] = *(const uint4*)&Plds[lm][16 + (hv << 3)];
    #pragma unroll
    for (int jd = 0; jd < 4; ++jd) {     // O += P * V
      Frag bv;
      const unsigned short* vr =
          v + rowbase + (size_t)(s0 + (hv << 3)) * E_ + (jd << 4) + lm;
      #pragma unroll
      for (int e = 0; e < 16; ++e) bv.s[e] = vr[(size_t)(e + (e & 8)) * E_];
      oacc[jd] = wmma_bf16(pa, bv, oacc[jd]);
    }
    __syncthreads();
  }

  #pragma unroll
  for (int jd = 0; jd < 4; ++jd)
    #pragma unroll
    for (int r = 0; r < 8; ++r) {
      float val = oacc[jd][r] / lsum[r];
      o[rowbase + (size_t)(q0 + r + (hv << 3)) * E_ + (jd << 4) + lm] = f2bf(val);
    }
}

// ---------------------------------------------------------------- driver
extern "C" void kernel_launch(void* const* d_in, const int* in_sizes, int n_in,
                              void* d_out, int out_size, void* d_ws, size_t ws_size,
                              hipStream_t stream) {
  (void)in_sizes; (void)n_in; (void)out_size; (void)ws_size;
  const int*   idx    = (const int*)d_in[0];
  const float* tok_w  = (const float*)d_in[1];
  const float* pos_w  = (const float*)d_in[2];
  const float* ln1_w  = (const float*)d_in[3];
  const float* ln1_b  = (const float*)d_in[4];
  const float* wq     = (const float*)d_in[5];
  const float* wk     = (const float*)d_in[6];
  const float* wv     = (const float*)d_in[7];
  const float* proj_w = (const float*)d_in[8];
  const float* proj_b = (const float*)d_in[9];
  const float* ln2_w  = (const float*)d_in[10];
  const float* ln2_b  = (const float*)d_in[11];
  const float* w1     = (const float*)d_in[12];
  const float* b1     = (const float*)d_in[13];
  const float* w2     = (const float*)d_in[14];
  const float* b2     = (const float*)d_in[15];
  const float* lnf_w  = (const float*)d_in[16];
  const float* lnf_b  = (const float*)d_in[17];
  const float* head_w = (const float*)d_in[18];
  const float* head_b = (const float*)d_in[19];

  char* ws = (char*)d_ws;
  float*          x  = (float*)ws;                              // 4096*768 f32
  unsigned short* xn = (unsigned short*)(ws + 12582912);        // bf16 LN out
  unsigned short* qb = (unsigned short*)(ws + 18874368);
  unsigned short* kb = (unsigned short*)(ws + 25165824);
  unsigned short* vb = (unsigned short*)(ws + 31457280);
  unsigned short* ob = (unsigned short*)(ws + 37748736);
  unsigned short* h1 = (unsigned short*)(ws + 44040192);        // 4096*3072 bf16

  embed_kernel<<<M_, 256, 0, stream>>>(idx, tok_w, pos_w, x);

  dim3 gE((E_ + 127) / 128, M_ / 128);    // 6 x 32
  dim3 gF((FF_ + 127) / 128, M_ / 128);   // 24 x 32
  dim3 gV((V_ + 127) / 128, M_ / 128);    // 393 x 32
  dim3 gA(T_ / 16, H_, B_);               // 64 x 12 x 4

  for (int l = 0; l < L_; ++l) {
    ln_kernel<<<M_, 256, 0, stream>>>(x, ln1_w + l * E_, ln1_b + l * E_, xn);
    gemm_kernel<true,  false, false, false><<<gE, 256, 0, stream>>>(
        xn, wq + (size_t)l * E_ * E_, nullptr, nullptr, qb, M_, E_, E_);
    gemm_kernel<true,  false, false, false><<<gE, 256, 0, stream>>>(
        xn, wk + (size_t)l * E_ * E_, nullptr, nullptr, kb, M_, E_, E_);
    gemm_kernel<true,  false, false, false><<<gE, 256, 0, stream>>>(
        xn, wv + (size_t)l * E_ * E_, nullptr, nullptr, vb, M_, E_, E_);
    attn_kernel<<<gA, 32, 0, stream>>>(qb, kb, vb, ob);
    gemm_kernel<false, true,  true,  false><<<gE, 256, 0, stream>>>(
        ob, proj_w + (size_t)l * E_ * E_, proj_b + l * E_, x, x, M_, E_, E_);
    ln_kernel<<<M_, 256, 0, stream>>>(x, ln2_w + l * E_, ln2_b + l * E_, xn);
    gemm_kernel<true,  true,  false, true ><<<gF, 256, 0, stream>>>(
        xn, w1 + (size_t)l * E_ * FF_, b1 + l * FF_, nullptr, h1, M_, FF_, E_);
    gemm_kernel<false, true,  true,  false><<<gE, 256, 0, stream>>>(
        h1, w2 + (size_t)l * FF_ * E_, b2 + l * E_, x, x, M_, E_, FF_);
  }
  ln_kernel<<<M_, 256, 0, stream>>>(x, lnf_w, lnf_b, xn);
  gemm_kernel<false, true,  false, false><<<gV, 256, 0, stream>>>(
      xn, head_w, head_b, nullptr, (float*)d_out, M_, V_, E_);
}
